// EdgeAwareGAT_89163521065370
// MI455X (gfx1250) — matrix operand order repaired and checked by reading, not compiled, not run on hardware
//
#include <hip/hip_runtime.h>
#include <hip/hip_bf16.h>

// ---------------------------------------------------------------------------
// EdgeAwareGAT on gfx1250 (MI455X, wave32).
// Dense GEMMs -> v_wmma_f32_16x16x32_f16 with LDS-staged, pre-swizzled f16
// B-fragments (ds_load_b128) and float4 A loads (global_load_b128).
// Edge softmax/aggregation -> encoded-uint atomicMax + global_atomic_add_f32.
// Fused ELU+residual+LayerNorm with wave32 shuffle reductions; pool + head.
// ---------------------------------------------------------------------------

typedef __attribute__((ext_vector_type(16))) _Float16 v16h;
typedef __attribute__((ext_vector_type(8)))  float    v8f;

#define WAVE 32

// ---------------- WMMA GEMM: Y = X[nrows,K] @ W[K,NC] (+bias) ---------------
// One wave computes a full 16-row stripe: 16 x NC output (NT = NC/16 tiles),
// reusing its A fragment across all column tiles. W is converted to f16 and
// staged in LDS once per block, already laid out per the ISA B-fragment
// layout (lane%16 = column i, lane half + element s = K-in-chunk), so a fragment fetch is a
// 32-byte contiguous LDS read.

template <int K, int NC>
__global__ __launch_bounds__(256)
void gemm_wmma_kernel(const float* __restrict__ X, const float* __restrict__ W,
                      const float* __restrict__ bias, float* __restrict__ Y,
                      int nrows) {
  constexpr int NT = NC / 16;  // column tiles per wave
  constexpr int KC = K / 32;   // K chunks
  __shared__ v16h Bfr[KC * NT * 32];

  // cooperative W(f32) -> LDS f16 fragment staging
  _Float16* bh = (_Float16*)Bfr;
  for (int idx = threadIdx.x; idx < K * NC; idx += blockDim.x) {
    const int k = idx / NC, n = idx % NC;
    const int kc = k >> 5, lkn = k & 31, ct = n >> 4, i = n & 15;
    const int L = (((lkn >> 3) & 1) << 4) + i, s = (lkn & 7) + ((lkn >> 4) << 3);
    bh[((((kc * NT + ct) << 5) + L) << 4) + s] = (_Float16)W[idx];
  }
  __syncthreads();

  const int rowTile = blockIdx.x * (blockDim.x >> 5) + (threadIdx.x >> 5);
  if (rowTile >= (nrows >> 4)) return;  // wave-uniform: EXEC all-1 for WMMA

  const int lane = threadIdx.x & 31;
  const int m    = lane & 15;
  const int g    = lane >> 4;  // lane half selects K sub-group
  const int row0 = rowTile << 4;

  v8f acc[NT];
#pragma unroll
  for (int t = 0; t < NT; ++t) acc[t] = (v8f){};

#pragma unroll
  for (int kc = 0; kc < KC; ++kc) {
    // A fragment: two contiguous 8-float runs per lane -> 4x float4 loads
    const float* rp = X + (size_t)(row0 + m) * K + (kc << 5);
    const float4 p0 = *(const float4*)(rp + 8 * g);
    const float4 p1 = *(const float4*)(rp + 8 * g + 4);
    const float4 p2 = *(const float4*)(rp + 16 + 8 * g);
    const float4 p3 = *(const float4*)(rp + 16 + 8 * g + 4);
    v16h a;
    a[0]  = (_Float16)p0.x; a[1]  = (_Float16)p0.y;
    a[2]  = (_Float16)p0.z; a[3]  = (_Float16)p0.w;
    a[4]  = (_Float16)p1.x; a[5]  = (_Float16)p1.y;
    a[6]  = (_Float16)p1.z; a[7]  = (_Float16)p1.w;
    a[8]  = (_Float16)p2.x; a[9]  = (_Float16)p2.y;
    a[10] = (_Float16)p2.z; a[11] = (_Float16)p2.w;
    a[12] = (_Float16)p3.x; a[13] = (_Float16)p3.y;
    a[14] = (_Float16)p3.z; a[15] = (_Float16)p3.w;
#pragma unroll
    for (int t = 0; t < NT; ++t) {
      const v16h b = Bfr[((kc * NT + t) << 5) + lane];  // 2x ds_load_b128
      acc[t] = __builtin_amdgcn_wmma_f32_16x16x32_f16(
          /*neg_a=*/false, a, /*neg_b=*/false, b,
          /*c_mod=*/(short)0, acc[t], /*reuse_a=*/false, /*reuse_b=*/false);
    }
  }

  const int n  = lane & 15;
  const int mo = g * 8;
#pragma unroll
  for (int t = 0; t < NT; ++t) {
    const float bv = bias ? bias[t * 16 + n] : 0.0f;
#pragma unroll
    for (int r = 0; r < 8; ++r)
      Y[(size_t)(row0 + mo + r) * NC + t * 16 + n] = acc[t][r] + bv;
  }
}

// ---------------- small utility kernels -------------------------------------

__global__ void fill_f32_kernel(float* p, size_t n, float v) {
  for (size_t i = blockIdx.x * (size_t)blockDim.x + threadIdx.x; i < n;
       i += (size_t)gridDim.x * blockDim.x)
    p[i] = v;
}
__global__ void fill_u32_kernel(unsigned* p, size_t n, unsigned v) {
  for (size_t i = blockIdx.x * (size_t)blockDim.x + threadIdx.x; i < n;
       i += (size_t)gridDim.x * blockDim.x)
    p[i] = v;
}

// order-preserving float <-> uint encoding for atomicMax-based segment max
__device__ __forceinline__ unsigned enc_f32(float f) {
  unsigned u = __float_as_uint(f);
  return (u & 0x80000000u) ? ~u : (u | 0x80000000u);
}
__device__ __forceinline__ float dec_f32(unsigned e) {
  unsigned u = (e & 0x80000000u) ? (e & 0x7fffffffu) : ~e;
  return __uint_as_float(u);
}

// per-(node,head): a_src/a_dst = <h[n,h,:], att_*[h,:]>
template <int H, int C>
__global__ void node_alpha_kernel(const float* __restrict__ hbuf,
                                  const float* __restrict__ att_src,
                                  const float* __restrict__ att_dst,
                                  float* __restrict__ a_src,
                                  float* __restrict__ a_dst, int n) {
  int t = blockIdx.x * blockDim.x + threadIdx.x;
  if (t >= n * H) return;
  const int node = t / H, h = t % H;
  const float* hp = hbuf + (size_t)node * (H * C) + h * C;
  float s = 0.f, d = 0.f;
#pragma unroll
  for (int c = 0; c < C; ++c) {
    s += hp[c] * att_src[h * C + c];
    d += hp[c] * att_dst[h * C + c];
  }
  a_src[t] = s;
  a_dst[t] = d;
}

// econst[h] = <W_e[0, h*C:...], att_e[h,:]>  (edge_attr is scalar per edge)
template <int H, int C>
__global__ void edge_const_kernel(const float* __restrict__ We,
                                  const float* __restrict__ att_e,
                                  float* __restrict__ econ) {
  int h = threadIdx.x;
  if (h >= H) return;
  float s = 0.f;
#pragma unroll
  for (int c = 0; c < C; ++c) s += We[h * C + c] * att_e[h * C + c];
  econ[h] = s;
}

template <int H>
__device__ __forceinline__ float edge_alpha(int u, int v, int h, float ea,
                                            const float* __restrict__ asrc,
                                            const float* __restrict__ adst,
                                            const float* __restrict__ econ) {
  float a = asrc[u * H + h] + adst[v * H + h] + ea * econ[h];
  return a > 0.f ? a : 0.2f * a;  // leaky_relu(0.2)
}

// pass 1: segment max over destination
template <int H>
__global__ void edge_max_kernel(const int* __restrict__ srci,
                                const int* __restrict__ dsti,
                                const float* __restrict__ eattr,
                                const float* __restrict__ asrc,
                                const float* __restrict__ adst,
                                const float* __restrict__ econ,
                                unsigned* __restrict__ mEnc, int nE) {
  const int tot = nE * H;
  for (int t = blockIdx.x * blockDim.x + threadIdx.x; t < tot;
       t += gridDim.x * blockDim.x) {
    const int e = t / H, h = t % H;
    const int u = srci[e], v = dsti[e];
    float a = edge_alpha<H>(u, v, h, eattr[e], asrc, adst, econ);
    atomicMax(&mEnc[v * H + h], enc_f32(a));
  }
}

// pass 2: segment sum of exp(alpha - max)
template <int H>
__global__ void edge_sum_kernel(const int* __restrict__ srci,
                                const int* __restrict__ dsti,
                                const float* __restrict__ eattr,
                                const float* __restrict__ asrc,
                                const float* __restrict__ adst,
                                const float* __restrict__ econ,
                                const unsigned* __restrict__ mEnc,
                                float* __restrict__ ssum, int nE) {
  const int tot = nE * H;
  for (int t = blockIdx.x * blockDim.x + threadIdx.x; t < tot;
       t += gridDim.x * blockDim.x) {
    const int e = t / H, h = t % H;
    const int u = srci[e], v = dsti[e];
    float a = edge_alpha<H>(u, v, h, eattr[e], asrc, adst, econ);
    atomicAdd(&ssum[v * H + h], __expf(a - dec_f32(mEnc[v * H + h])));
  }
}

// pass 3: weighted scatter-add of messages
template <int H, int C>
__global__ void edge_aggr_kernel(const int* __restrict__ srci,
                                 const int* __restrict__ dsti,
                                 const float* __restrict__ eattr,
                                 const float* __restrict__ asrc,
                                 const float* __restrict__ adst,
                                 const float* __restrict__ econ,
                                 const unsigned* __restrict__ mEnc,
                                 const float* __restrict__ ssum,
                                 const float* __restrict__ hbuf,
                                 float* __restrict__ aggr, int nE) {
  const int tot = nE * H;
  for (int t = blockIdx.x * blockDim.x + threadIdx.x; t < tot;
       t += gridDim.x * blockDim.x) {
    const int e = t / H, h = t % H;
    const int u = srci[e], v = dsti[e];
    float a = edge_alpha<H>(u, v, h, eattr[e], asrc, adst, econ);
    const int vh = v * H + h;
    const float coef = __expf(a - dec_f32(mEnc[vh])) / (ssum[vh] + 1e-16f);
    const float* hp = hbuf + (size_t)u * (H * C) + h * C;
    float* op = aggr + (size_t)v * (H * C) + h * C;
    __builtin_prefetch(hp, 0, 0);  // global_prefetch_b8
#pragma unroll
    for (int c = 0; c < C; ++c) atomicAdd(&op[c], hp[c] * coef);
  }
}

// fused: out = LayerNorm( res + elu(aggr + bias) ) * gamma + beta
// one wave per node, D in {32, 64}
template <int D>
__global__ __launch_bounds__(256)
void fuse_ln_kernel(const float* __restrict__ aggr, const float* __restrict__ bias,
                    const float* __restrict__ res, const float* __restrict__ gamma,
                    const float* __restrict__ beta, float* __restrict__ out, int n) {
  const int w = (int)((blockIdx.x * blockDim.x + threadIdx.x) >> 5);
  const int lane = threadIdx.x & 31;
  if (w >= n) return;
  constexpr int F = D / 32;
  float v[F];
#pragma unroll
  for (int f = 0; f < F; ++f) {
    const int c = f * 32 + lane;
    float x = aggr[(size_t)w * D + c] + bias[c];
    x = x > 0.f ? x : (__expf(x) - 1.f);  // ELU
    v[f] = x + res[(size_t)w * D + c];
  }
  float sum = 0.f;
#pragma unroll
  for (int f = 0; f < F; ++f) sum += v[f];
#pragma unroll
  for (int o = 16; o > 0; o >>= 1) sum += __shfl_xor(sum, o, 32);
  const float mu = sum / (float)D;
  float var = 0.f;
#pragma unroll
  for (int f = 0; f < F; ++f) { float d = v[f] - mu; var += d * d; }
#pragma unroll
  for (int o = 16; o > 0; o >>= 1) var += __shfl_xor(var, o, 32);
  const float rs = rsqrtf(var / (float)D + 1e-5f);
#pragma unroll
  for (int f = 0; f < F; ++f) {
    const int c = f * 32 + lane;
    out[(size_t)w * D + c] = (v[f] - mu) * rs * gamma[c] + beta[c];
  }
}

// mean-pool accumulation: one wave per node, 32 channels
__global__ __launch_bounds__(256)
void pool_kernel(const float* __restrict__ h, const int* __restrict__ batch,
                 float* __restrict__ pooled, float* __restrict__ cnt, int n) {
  const int w = (int)((blockIdx.x * blockDim.x + threadIdx.x) >> 5);
  const int lane = threadIdx.x & 31;
  if (w >= n) return;
  const int b = batch[w];
  atomicAdd(&pooled[b * 32 + lane], h[(size_t)w * 32 + lane]);
  if (lane == 0) atomicAdd(&cnt[b], 1.0f);
}

__global__ void head_kernel(const float* __restrict__ pooled,
                            const float* __restrict__ cnt,
                            const float* __restrict__ wout,
                            const float* __restrict__ bout,
                            float* __restrict__ out, int G) {
  int g = blockIdx.x * blockDim.x + threadIdx.x;
  if (g >= G) return;
  const float c = fmaxf(cnt[g], 1.0f);
  float s = 0.f;
#pragma unroll
  for (int i = 0; i < 32; ++i) s += (pooled[g * 32 + i] / c) * wout[i];
  out[g] = s + bout[0];
}

// ---------------------------------------------------------------------------

static inline int cdiv(long long a, long long b) { return (int)((a + b - 1) / b); }

extern "C" void kernel_launch(void* const* d_in, const int* in_sizes, int n_in,
                              void* d_out, int out_size, void* d_ws, size_t ws_size,
                              hipStream_t stream) {
  // ---- inputs (setup_inputs dict order) ----
  const float* x        = (const float*)d_in[0];
  const int*   ei       = (const int*)d_in[1];
  const float* eattr    = (const float*)d_in[2];
  const int*   batch    = (const int*)d_in[3];
  const float* lin_in_w = (const float*)d_in[4];
  const float* lin_in_b = (const float*)d_in[5];
  const float* g1_w     = (const float*)d_in[6];
  const float* g1_asrc  = (const float*)d_in[7];
  const float* g1_adst  = (const float*)d_in[8];
  const float* g1_We    = (const float*)d_in[9];
  const float* g1_ae    = (const float*)d_in[10];
  const float* g1_bias  = (const float*)d_in[11];
  const float* n1_g     = (const float*)d_in[12];
  const float* n1_b     = (const float*)d_in[13];
  const float* g2_w     = (const float*)d_in[14];
  const float* g2_asrc  = (const float*)d_in[15];
  const float* g2_adst  = (const float*)d_in[16];
  const float* g2_We    = (const float*)d_in[17];
  const float* g2_ae    = (const float*)d_in[18];
  const float* g2_bias  = (const float*)d_in[19];
  const float* lres_w   = (const float*)d_in[20];
  const float* lres_b   = (const float*)d_in[21];
  const float* n2_g     = (const float*)d_in[22];
  const float* n2_b     = (const float*)d_in[23];
  const float* g3_w     = (const float*)d_in[24];
  const float* g3_asrc  = (const float*)d_in[25];
  const float* g3_adst  = (const float*)d_in[26];
  const float* g3_We    = (const float*)d_in[27];
  const float* g3_ae    = (const float*)d_in[28];
  const float* g3_bias  = (const float*)d_in[29];
  const float* n3_g     = (const float*)d_in[30];
  const float* n3_b     = (const float*)d_in[31];
  const float* lout_w   = (const float*)d_in[32];
  const float* lout_b   = (const float*)d_in[33];

  const int N = in_sizes[3];  // batch vector length (nodes)
  const int E = in_sizes[2];  // edge_attr length (edges)
  const int G = out_size;     // graphs
  const int* srci = ei;
  const int* dsti = ei + E;

  // ---- workspace carve-up (floats) ----
  float* ws = (float*)d_ws;
  const size_t N64 = (size_t)N * 64, N32 = (size_t)N * 32, N2 = (size_t)N * 2;
  float*    x0     = ws;                      // N x 64
  float*    hbuf   = x0   + N64;              // N x 64 (per-layer h = x @ W)
  float*    aggr   = hbuf + N64;              // N x 64 (message accumulator)
  float*    hcur   = aggr + N64;              // N x 64 (running node state)
  float*    xres   = hcur + N64;              // N x 32
  float*    asrc   = xres + N32;              // N x 2
  float*    adst   = asrc + N2;               // N x 2
  unsigned* mEnc   = (unsigned*)(adst + N2);  // N x 2
  float*    ssum   = (float*)mEnc + N2;       // N x 2
  float*    econ   = ssum + N2;               // 2
  float*    pooled = econ + 2;                // G x 32
  float*    cnt    = pooled + (size_t)G * 32; // G
  float*    outF   = (float*)d_out;

  const int TB = 256;
  const int fillB = 2048;
  const int edgeB = cdiv((long long)E * 2, TB) > 16384 ? 16384
                                                       : cdiv((long long)E * 2, TB);
  const int nodeWaveB = cdiv(N, TB / WAVE);
  const int gemmB     = cdiv(N / 16, TB / WAVE);  // one wave per 16-row stripe

  // ===================== lin_in: x0 = x @ W + b  [N,128]->[N,64] ============
  gemm_wmma_kernel<128, 64><<<gemmB, TB, 0, stream>>>(x, lin_in_w, lin_in_b, x0, N);

  // ===================== layer 1: GAT(64->32, H=2, concat) ==================
  gemm_wmma_kernel<64, 64><<<gemmB, TB, 0, stream>>>(x0, g1_w, nullptr, hbuf, N);
  node_alpha_kernel<2, 32><<<cdiv((long long)N * 2, TB), TB, 0, stream>>>(
      hbuf, g1_asrc, g1_adst, asrc, adst, N);
  edge_const_kernel<2, 32><<<1, 32, 0, stream>>>(g1_We, g1_ae, econ);
  fill_u32_kernel<<<fillB, TB, 0, stream>>>(mEnc, N2, 0u);
  fill_f32_kernel<<<fillB, TB, 0, stream>>>(ssum, N2, 0.f);
  fill_f32_kernel<<<fillB, TB, 0, stream>>>(aggr, N64, 0.f);
  edge_max_kernel<2><<<edgeB, TB, 0, stream>>>(srci, dsti, eattr, asrc, adst, econ, mEnc, E);
  edge_sum_kernel<2><<<edgeB, TB, 0, stream>>>(srci, dsti, eattr, asrc, adst, econ, mEnc, ssum, E);
  edge_aggr_kernel<2, 32><<<edgeB, TB, 0, stream>>>(srci, dsti, eattr, asrc, adst,
                                                    econ, mEnc, ssum, hbuf, aggr, E);
  fuse_ln_kernel<64><<<nodeWaveB, TB, 0, stream>>>(aggr, g1_bias, x0, n1_g, n1_b, hcur, N);

  // ===================== layer 2: GAT(64->32, H=1) + lin_res ================
  gemm_wmma_kernel<64, 32><<<gemmB, TB, 0, stream>>>(hcur, g2_w, nullptr, hbuf, N);
  gemm_wmma_kernel<64, 32><<<gemmB, TB, 0, stream>>>(hcur, lres_w, lres_b, xres, N);
  node_alpha_kernel<1, 32><<<cdiv(N, TB), TB, 0, stream>>>(
      hbuf, g2_asrc, g2_adst, asrc, adst, N);
  edge_const_kernel<1, 32><<<1, 32, 0, stream>>>(g2_We, g2_ae, econ);
  fill_u32_kernel<<<fillB, TB, 0, stream>>>(mEnc, (size_t)N, 0u);
  fill_f32_kernel<<<fillB, TB, 0, stream>>>(ssum, (size_t)N, 0.f);
  fill_f32_kernel<<<fillB, TB, 0, stream>>>(aggr, N32, 0.f);
  edge_max_kernel<1><<<edgeB, TB, 0, stream>>>(srci, dsti, eattr, asrc, adst, econ, mEnc, E);
  edge_sum_kernel<1><<<edgeB, TB, 0, stream>>>(srci, dsti, eattr, asrc, adst, econ, mEnc, ssum, E);
  edge_aggr_kernel<1, 32><<<edgeB, TB, 0, stream>>>(srci, dsti, eattr, asrc, adst,
                                                    econ, mEnc, ssum, hbuf, aggr, E);
  fuse_ln_kernel<32><<<nodeWaveB, TB, 0, stream>>>(aggr, g2_bias, xres, n2_g, n2_b, hcur, N);

  // ===================== layer 3: GAT(32->32, H=1) ==========================
  gemm_wmma_kernel<32, 32><<<gemmB, TB, 0, stream>>>(hcur, g3_w, nullptr, hbuf, N);
  node_alpha_kernel<1, 32><<<cdiv(N, TB), TB, 0, stream>>>(
      hbuf, g3_asrc, g3_adst, asrc, adst, N);
  edge_const_kernel<1, 32><<<1, 32, 0, stream>>>(g3_We, g3_ae, econ);
  fill_u32_kernel<<<fillB, TB, 0, stream>>>(mEnc, (size_t)N, 0u);
  fill_f32_kernel<<<fillB, TB, 0, stream>>>(ssum, (size_t)N, 0.f);
  fill_f32_kernel<<<fillB, TB, 0, stream>>>(aggr, N32, 0.f);
  edge_max_kernel<1><<<edgeB, TB, 0, stream>>>(srci, dsti, eattr, asrc, adst, econ, mEnc, E);
  edge_sum_kernel<1><<<edgeB, TB, 0, stream>>>(srci, dsti, eattr, asrc, adst, econ, mEnc, ssum, E);
  edge_aggr_kernel<1, 32><<<edgeB, TB, 0, stream>>>(srci, dsti, eattr, asrc, adst,
                                                    econ, mEnc, ssum, hbuf, aggr, E);
  // residual = hcur (read-before-write per element, safe in-place)
  fuse_ln_kernel<32><<<nodeWaveB, TB, 0, stream>>>(aggr, g3_bias, hcur, n3_g, n3_b, hcur, N);

  // ===================== mean-pool per graph + head =========================
  fill_f32_kernel<<<32, TB, 0, stream>>>(pooled, (size_t)G * 32, 0.f);
  fill_f32_kernel<<<1, TB, 0, stream>>>(cnt, (size_t)G, 0.f);
  pool_kernel<<<nodeWaveB, TB, 0, stream>>>(hcur, batch, pooled, cnt, N);
  head_kernel<<<1, 64, 0, stream>>>(pooled, cnt, lout_w, lout_b, outF, G);
}
